// BasicBlock_13752485281975
// MI455X (gfx1250) — matrix operand order repaired and checked by reading, not compiled, not run on hardware
//
#include <hip/hip_runtime.h>
#include <math.h>

// ---------------------------------------------------------------------------
// BasicBlock: ternary-quantized 3x3 conv (+doubled center tap) -> BN -> ReLU,
// twice, with residual add.  Convs run as implicit GEMM on v_wmma_f32_16x16x32_f16
// with NHWC f16 activations and (where available) async global->LDS DMA.
// ---------------------------------------------------------------------------

typedef __attribute__((ext_vector_type(16))) _Float16 v16h;
typedef __attribute__((ext_vector_type(8)))  _Float16 v8h;
typedef __attribute__((ext_vector_type(8)))  float    v8f;

#define C_IN   256
#define C_OUT  256
#define IMG_H  56
#define IMG_W  56
#define HW     (IMG_H * IMG_W)           /* 3136  */
#define NIMG   16
#define STOT   (NIMG * HW)               /* 50176 */
#define KTOT   (C_IN * 9)                /* 2304  */
#define WELEM  (C_OUT * C_IN * 9)        /* 589824 */

#define BM 64      /* block tile: output channels   */
#define BN 128     /* block tile: spatial positions */
#define BK 32      /* K step (matches WMMA K=32)    */
#define LDAH 40    /* A row stride in halves (32 + 8 pad -> 80B, 16B aligned) */
#define LDBH 40    /* B row stride in halves */
#define NKSTEP (KTOT / BK)   /* 72 */

#if __has_builtin(__builtin_amdgcn_global_load_async_to_lds_b128) && \
    __has_builtin(__builtin_amdgcn_s_wait_asynccnt)
#define USE_ASYNC_LDS 1
#else
#define USE_ASYNC_LDS 0
#endif

#if USE_ASYNC_LDS
// Builtin signature (from hipcc diagnostic): param0 = int4 in AS1 ("__device__"),
// param1 = int4 in AS3, then immediate offset + cpol.  Flat shared pointers
// carry the LDS offset in their low 32 bits (aperture layout), so integer casts
// materialize the AS-qualified pointers.
typedef int v4i_g __attribute__((vector_size(16)));
typedef __attribute__((address_space(1))) v4i_g* as1_v4i_ptr;
typedef __attribute__((address_space(3))) v4i_g* as3_v4i_ptr;
static __device__ inline void async_copy16(const void* g, void* l) {
    __builtin_amdgcn_global_load_async_to_lds_b128(
        (as1_v4i_ptr)(unsigned long long)(size_t)g,
        (as3_v4i_ptr)(unsigned int)(size_t)l, 0, 0);
}
#endif

static __device__ inline v16h cat8(v8h a, v8h b) {
    return __builtin_shufflevector(a, b, 0,1,2,3,4,5,6,7,8,9,10,11,12,13,14,15);
}

// ---------------------------------------------------------------------------
// Weight stats: per-block partial {sum, max, min} (deterministic tree reduce)
// ---------------------------------------------------------------------------
__global__ __launch_bounds__(256)
void wreduce_kernel(const float* __restrict__ W, int n, float* __restrict__ part) {
    __shared__ float rs[256], rmx[256], rmn[256];
    const int tid = threadIdx.x;
    float s = 0.f, mx = -__builtin_huge_valf(), mn = __builtin_huge_valf();
    for (int i = blockIdx.x * 256 + tid; i < n; i += gridDim.x * 256) {
        float v = W[i];
        s += v; mx = fmaxf(mx, v); mn = fminf(mn, v);
    }
    rs[tid] = s; rmx[tid] = mx; rmn[tid] = mn;
    __syncthreads();
    for (int off = 128; off > 0; off >>= 1) {
        if (tid < off) {
            rs[tid]  += rs[tid + off];
            rmx[tid]  = fmaxf(rmx[tid], rmx[tid + off]);
            rmn[tid]  = fminf(rmn[tid], rmn[tid + off]);
        }
        __syncthreads();
    }
    if (tid == 0) {
        part[blockIdx.x * 3 + 0] = rs[0];
        part[blockIdx.x * 3 + 1] = rmx[0];
        part[blockIdx.x * 3 + 2] = rmn[0];
    }
}

// Finalize: stats = {mean, lo, hi} thresholds on (w - mean)
__global__ __launch_bounds__(64)
void wfinal_kernel(const float* __restrict__ part, int nblk, float inv_n,
                   float* __restrict__ stats) {
    __shared__ float rs[64], rmx[64], rmn[64];
    const int t = threadIdx.x;
    float s = 0.f, mx = -__builtin_huge_valf(), mn = __builtin_huge_valf();
    if (t < nblk) { s = part[t*3]; mx = part[t*3+1]; mn = part[t*3+2]; }
    rs[t] = s; rmx[t] = mx; rmn[t] = mn;
    __syncthreads();
    for (int off = 32; off > 0; off >>= 1) {
        if (t < off) {
            rs[t]  += rs[t + off];
            rmx[t]  = fmaxf(rmx[t], rmx[t + off]);
            rmn[t]  = fminf(rmn[t], rmn[t + off]);
        }
        __syncthreads();
    }
    if (t == 0) {
        float mean = rs[0] * inv_n;
        float mxt = rmx[0] - mean;
        float mnt = rmn[0] - mean;
        float d = (mxt - mnt) * (1.f / 3.f);
        stats[0] = mean;
        stats[1] = mnt + d;               // lo
        stats[2] = mxt - d;               // hi
    }
}

// Quantize OIHW f32 -> f16 ternary with doubled center, layout [co][(kh*3+kw)*256+ci]
__global__ __launch_bounds__(256)
void wquant_kernel(const float* __restrict__ W, const float* __restrict__ stats,
                   _Float16* __restrict__ Wq) {
    const int idx = blockIdx.x * 256 + threadIdx.x;
    if (idx >= WELEM) return;
    const int co  = idx / KTOT;
    const int k   = idx - co * KTOT;
    const int khw = k >> 8;
    const int ci  = k & 255;
    const int kh  = khw / 3, kw = khw - kh * 3;
    const float mean = stats[0], lo = stats[1], hi = stats[2];
    const float tw = W[((co * C_IN + ci) * 3 + kh) * 3 + kw] - mean;
    float q = (tw < lo) ? -1.f : ((tw > hi) ? 1.f : 0.f);
    if (kh == 1 && kw == 1) q *= 2.f;     // merged 1x1 "extra" term
    Wq[idx] = (_Float16)q;
}

// NCHW f32 -> NHWC f16 (conv B-operand layout: channel contiguous per pixel)
__global__ __launch_bounds__(256)
void nhwc_cvt_kernel(const float* __restrict__ X, _Float16* __restrict__ Xh) {
    const long total = (long)NIMG * HW * C_IN;
    const long stride = (long)gridDim.x * 256;
    for (long i = (long)blockIdx.x * 256 + threadIdx.x; i < total; i += stride) {
        const int  c  = (int)(i & (C_IN - 1));
        const long np = i >> 8;                  // n*HW + p
        const long n  = np / HW;
        const long p  = np - n * HW;
        Xh[i] = (_Float16)X[(n * C_IN + c) * HW + p];
    }
}

// ---------------------------------------------------------------------------
// Implicit-GEMM conv.  Xh is NHWC f16; Wq is f16 [co][K]; Out is NCHW f32 raw.
// Block: 256 threads / 8 waves, tile 64(M) x 128(N), K step 32, double-buffered
// LDS filled by async global->LDS DMA (zero ds-stores on border lanes).
// ---------------------------------------------------------------------------
__global__ __launch_bounds__(256)
void conv_wmma_kernel(const _Float16* __restrict__ Xh,
                      const _Float16* __restrict__ Wq,
                      float* __restrict__ Out) {
    __shared__ __align__(16) _Float16 Alds[2][BM * LDAH];   // 2 x 5120 B
    __shared__ __align__(16) _Float16 Blds[2][BN * LDBH];   // 2 x 10240 B

    const int tid  = threadIdx.x;
    const int lane = tid & 31;
    const int wv   = tid >> 5;
    const int wm   = wv & 1;            // 32-row M half
    const int wn   = wv >> 1;           // 32-col N quarter
    const int co0  = blockIdx.y * BM;
    const int s0   = blockIdx.x * BN;

    // B-tile role: one spatial position + 16-channel subgroup per thread.
    const int sl   = tid & 127;
    const int cig  = tid >> 7;
    const int sg   = s0 + sl;
    const int nimg = sg / HW;
    const int rp   = sg - nimg * HW;
    const int hh   = rp / IMG_W;
    const int ww   = rp - hh * IMG_W;

    // A-tile role: 16B of one weight row per thread.
    const int am  = tid >> 2;           // 0..63
    const int ak8 = (tid & 3) * 8;      // 0,8,16,24

    auto issue_tile = [&](int kk, int buf) {
        const int khw = kk >> 3;                  // (kh,kw) group 0..8
        const int kh  = khw / 3, kw = khw - kh * 3;
        const int ci0 = (kk & 7) * BK;
        // ---- B: 32 ci x 128 spatial, NHWC so 2x16B per thread
        const int py = hh + kh - 1;
        const int px = ww + kw - 1;
        const bool valid = (py >= 0) && (py < IMG_H) && (px >= 0) && (px < IMG_W);
        _Float16* bdst = &Blds[buf][sl * LDBH + cig * 16];
        if (valid) {
            const _Float16* gsrc =
                Xh + ((long)nimg * HW + (long)py * IMG_W + px) * C_IN + ci0 + cig * 16;
#if USE_ASYNC_LDS
            async_copy16(gsrc, bdst);
            async_copy16(gsrc + 8, bdst + 8);
#else
            ((v8h*)bdst)[0] = *(const v8h*)gsrc;
            ((v8h*)bdst)[1] = *(const v8h*)(gsrc + 8);
#endif
        } else {
            v8h z;
            #pragma unroll
            for (int i = 0; i < 8; ++i) z[i] = (_Float16)0.f;
            ((v8h*)bdst)[0] = z;
            ((v8h*)bdst)[1] = z;
        }
        // ---- A: 64 co x 32 K, contiguous rows, 16B per thread
        const _Float16* asrc = Wq + (long)(co0 + am) * KTOT + kk * BK + ak8;
        _Float16* adst = &Alds[buf][am * LDAH + ak8];
#if USE_ASYNC_LDS
        async_copy16(asrc, adst);
#else
        *(v8h*)adst = *(const v8h*)asrc;
#endif
    };

    v8f acc[2][2];
    #pragma unroll
    for (int i = 0; i < 2; ++i)
        #pragma unroll
        for (int j = 0; j < 2; ++j)
            #pragma unroll
            for (int r = 0; r < 8; ++r) acc[i][j][r] = 0.f;

    issue_tile(0, 0);
#if USE_ASYNC_LDS
    __builtin_amdgcn_s_wait_asynccnt(0);
#endif
    __syncthreads();

    const int half = lane >> 4;
    const int l16  = lane & 15;

    for (int kk = 0; kk < NKSTEP; ++kk) {
        const int cur = kk & 1;
        if (kk + 1 < NKSTEP) issue_tile(kk + 1, cur ^ 1);

        // ---- fragments in contiguous K order (k = 16*half + s, A and B alike), then 2x2 WMMA on current buffer
        v16h afrag[2], bfrag[2];
        #pragma unroll
        for (int mt = 0; mt < 2; ++mt) {
            const _Float16* row = &Alds[cur][(wm * 32 + mt * 16 + l16) * LDAH];
            v8h a0 = *(const v8h*)&row[half * 16];       // K = 16*half + 0..7
            v8h a1 = *(const v8h*)&row[half * 16 + 8];   // K = 16*half + 8..15
            afrag[mt] = cat8(a0, a1);
        }
        #pragma unroll
        for (int nt = 0; nt < 2; ++nt) {
            const _Float16* row = &Blds[cur][(wn * 32 + nt * 16 + l16) * LDBH];
            v8h b0 = *(const v8h*)&row[half * 16];       // K = 16*half + 0..7
            v8h b1 = *(const v8h*)&row[half * 16 + 8];   // K = 16*half + 8..15
            bfrag[nt] = cat8(b0, b1);
        }
        #pragma unroll
        for (int mt = 0; mt < 2; ++mt)
            #pragma unroll
            for (int nt = 0; nt < 2; ++nt)
                acc[mt][nt] = __builtin_amdgcn_wmma_f32_16x16x32_f16(
                    false, afrag[mt], false, bfrag[nt],
                    (short)0, acc[mt][nt], false, false);

#if USE_ASYNC_LDS
        if (kk + 1 < NKSTEP) __builtin_amdgcn_s_wait_asynccnt(0);
#endif
        __syncthreads();
    }

    // ---- store raw f32 conv output, NCHW (C/D layout: VGPR r -> M = 8*half + r)
    #pragma unroll
    for (int mt = 0; mt < 2; ++mt) {
        const int cob = co0 + wm * 32 + mt * 16 + half * 8;
        #pragma unroll
        for (int nt = 0; nt < 2; ++nt) {
            const int s = s0 + wn * 32 + nt * 16 + l16;
            const int n = s / HW;
            const int p = s - n * HW;
            float* outp = Out + (long)n * (C_OUT * HW) + (long)cob * HW + p;
            #pragma unroll
            for (int r = 0; r < 8; ++r)
                outp[(long)r * HW] = acc[mt][nt][r];
        }
    }
}

// ---------------------------------------------------------------------------
// BN statistics: one block per channel, deterministic tree reduce.
// ---------------------------------------------------------------------------
__global__ __launch_bounds__(256)
void bnstats_kernel(const float* __restrict__ Y, float* __restrict__ sum,
                    float* __restrict__ sumsq) {
    __shared__ float rs[256], rq[256];
    const int c = blockIdx.x;
    const int tid = threadIdx.x;
    float s = 0.f, q = 0.f;
    for (int n = 0; n < NIMG; ++n) {
        const float* base = Y + (long)n * (C_OUT * HW) + (long)c * HW;
        for (int p = tid; p < HW; p += 256) {
            float v = base[p];
            s += v; q += v * v;
        }
    }
    rs[tid] = s; rq[tid] = q;
    __syncthreads();
    for (int off = 128; off > 0; off >>= 1) {
        if (tid < off) { rs[tid] += rs[tid + off]; rq[tid] += rq[tid + off]; }
        __syncthreads();
    }
    if (tid == 0) { sum[c] = rs[0]; sumsq[c] = rq[0]; }
}

// BN + ReLU (block 1): writes f32 NCHW (residual) and f16 NHWC (conv2 input)
__global__ __launch_bounds__(256)
void bnrelu_kernel(const float* __restrict__ raw, const float* __restrict__ sum,
                   const float* __restrict__ sumsq, const float* __restrict__ gamma,
                   const float* __restrict__ beta, float* __restrict__ out,
                   _Float16* __restrict__ outh) {
    const long total = (long)NIMG * C_OUT * HW;
    const long stride = (long)gridDim.x * 256;
    for (long i = (long)blockIdx.x * 256 + threadIdx.x; i < total; i += stride) {
        const int  c = (int)((i / HW) & (C_OUT - 1));
        const long n = i / ((long)C_OUT * HW);
        const long p = i - (n * C_OUT + c) * HW;
        const float mean = sum[c] * (1.f / STOT);
        const float var  = sumsq[c] * (1.f / STOT) - mean * mean;
        const float y = (raw[i] - mean) * rsqrtf(var + 1e-5f) * gamma[c] + beta[c];
        const float a = fmaxf(y, 0.f);
        out[i] = a;
        outh[(n * HW + p) * C_IN + c] = (_Float16)a;
    }
}

// Final: out = relu(act1 + bn2(raw2))
__global__ __launch_bounds__(256)
void bnfinal_kernel(const float* __restrict__ raw, const float* __restrict__ sum,
                    const float* __restrict__ sumsq, const float* __restrict__ gamma,
                    const float* __restrict__ beta, const float* __restrict__ act1,
                    float* __restrict__ out) {
    const long total = (long)NIMG * C_OUT * HW;
    const long stride = (long)gridDim.x * 256;
    for (long i = (long)blockIdx.x * 256 + threadIdx.x; i < total; i += stride) {
        const int c = (int)((i / HW) & (C_OUT - 1));
        const float mean = sum[c] * (1.f / STOT);
        const float var  = sumsq[c] * (1.f / STOT) - mean * mean;
        const float y = (raw[i] - mean) * rsqrtf(var + 1e-5f) * gamma[c] + beta[c];
        out[i] = fmaxf(act1[i] + y, 0.f);
    }
}

// ---------------------------------------------------------------------------
extern "C" void kernel_launch(void* const* d_in, const int* in_sizes, int n_in,
                              void* d_out, int out_size, void* d_ws, size_t ws_size,
                              hipStream_t stream) {
    const float* x  = (const float*)d_in[0];
    const float* w1 = (const float*)d_in[1];
    const float* w2 = (const float*)d_in[2];
    const float* g1 = (const float*)d_in[3];
    const float* b1 = (const float*)d_in[4];
    const float* g2 = (const float*)d_in[5];
    const float* b2 = (const float*)d_in[6];
    float* out = (float*)d_out;

    const long NELEM = (long)NIMG * C_OUT * HW;   // 12,845,056

    // Workspace carve (256B aligned).  bufRaw ping-pongs between the two convs;
    // bufH holds x(NHWC f16) for conv1, then is overwritten with act1(NHWC f16)
    // for conv2.
    char* p = (char*)d_ws;
    auto take = [&](size_t bytes) -> void* {
        void* r = (void*)p;
        p += (bytes + 255) & ~(size_t)255;
        return r;
    };
    float*     bufRaw  = (float*)take(NELEM * sizeof(float));
    float*     bufAct  = (float*)take(NELEM * sizeof(float));
    _Float16*  bufH    = (_Float16*)take(NELEM * sizeof(_Float16));
    _Float16*  Wq1     = (_Float16*)take((size_t)C_OUT * KTOT * 2);
    _Float16*  Wq2     = (_Float16*)take((size_t)C_OUT * KTOT * 2);
    float*     part1   = (float*)take(64 * 3 * sizeof(float));
    float*     part2   = (float*)take(64 * 3 * sizeof(float));
    float*     wstats1 = (float*)take(3 * sizeof(float));
    float*     wstats2 = (float*)take(3 * sizeof(float));
    float*     sum1    = (float*)take(C_OUT * sizeof(float));
    float*     sq1     = (float*)take(C_OUT * sizeof(float));
    float*     sum2    = (float*)take(C_OUT * sizeof(float));
    float*     sq2     = (float*)take(C_OUT * sizeof(float));

    const float inv_w = 1.f / (float)WELEM;

    // --- quantize both weight tensors; convert x to NHWC f16
    wreduce_kernel<<<64, 256, 0, stream>>>(w1, WELEM, part1);
    wreduce_kernel<<<64, 256, 0, stream>>>(w2, WELEM, part2);
    wfinal_kernel<<<1, 64, 0, stream>>>(part1, 64, inv_w, wstats1);
    wfinal_kernel<<<1, 64, 0, stream>>>(part2, 64, inv_w, wstats2);
    wquant_kernel<<<(WELEM + 255) / 256, 256, 0, stream>>>(w1, wstats1, Wq1);
    wquant_kernel<<<(WELEM + 255) / 256, 256, 0, stream>>>(w2, wstats2, Wq2);
    nhwc_cvt_kernel<<<4096, 256, 0, stream>>>(x, bufH);

    const dim3 cgrid(STOT / BN, C_OUT / BM, 1);   // 392 x 4

    // --- block 1
    conv_wmma_kernel<<<cgrid, 256, 0, stream>>>(bufH, Wq1, bufRaw);
    bnstats_kernel<<<C_OUT, 256, 0, stream>>>(bufRaw, sum1, sq1);
    bnrelu_kernel<<<4096, 256, 0, stream>>>(bufRaw, sum1, sq1, g1, b1, bufAct, bufH);

    // --- block 2
    conv_wmma_kernel<<<cgrid, 256, 0, stream>>>(bufH, Wq2, bufRaw);
    bnstats_kernel<<<C_OUT, 256, 0, stream>>>(bufRaw, sum2, sq2);
    bnfinal_kernel<<<4096, 256, 0, stream>>>(bufRaw, sum2, sq2, g2, b2, bufAct, out);
}
